// Conv2d_int8_34626026340864
// MI455X (gfx1250) — 8-bit matrix operand order repaired and checked by reading, compiled, not run on hardware
//
#include <hip/hip_runtime.h>
#include <hip/hip_bf16.h>

typedef __attribute__((ext_vector_type(8))) int v8i;

#define N_BATCH   32
#define C_IN      32
#define H_IN      160
#define W_IN      160
#define O_OUT     32
#define HP        162
#define WP        162
#define HW        (H_IN * W_IN)
#define NPIX_PAD  (N_BATCH * HP * WP)
#define NX        (N_BATCH * C_IN * HW)
#define NW        (O_OUT * C_IN * 9)

#define WQ_OFF    256
#define XQ_OFF    16384

__global__ void zero_amax_kernel(unsigned* amax) {
    if (threadIdx.x < 2) amax[threadIdx.x] = 0u;
}

__global__ void absmax_kernel(const float* __restrict__ v, int n,
                              unsigned* __restrict__ out) {
    __shared__ unsigned sm[256];
    float m = 0.0f;
    for (int i = blockIdx.x * blockDim.x + threadIdx.x; i < n;
         i += gridDim.x * blockDim.x)
        m = fmaxf(m, fabsf(v[i]));
    sm[threadIdx.x] = __float_as_uint(m);
    __syncthreads();
    for (int s = 128; s > 0; s >>= 1) {
        if (threadIdx.x < s)
            sm[threadIdx.x] = max(sm[threadIdx.x], sm[threadIdx.x + s]);
        __syncthreads();
    }
    if (threadIdx.x == 0) atomicMax(out, sm[0]);
}

__device__ __forceinline__ float qd_scale(unsigned bits) {
    return fmaxf(__uint_as_float(bits) * (1.0f / 127.0f), 1e-12f);
}

__device__ __forceinline__ int q8(float v, float inv_scale) {
    int q = (int)rintf(v * inv_scale);
    q = q > 127 ? 127 : (q < -127 ? -127 : q);
    return q & 0xFF;
}

__global__ void quantize_x_kernel(const float* __restrict__ x,
                                  const unsigned* __restrict__ amax,
                                  signed char* __restrict__ xq) {
    int pid = blockIdx.x * blockDim.x + threadIdx.x;
    if (pid >= NPIX_PAD) return;
    int n  = pid / (HP * WP);
    int r  = pid % (HP * WP);
    int yy = r / WP;
    int xx = r % WP;
    int* dst = (int*)(xq + (long)pid * 32);
    if (yy == 0 || yy == HP - 1 || xx == 0 || xx == WP - 1) {
#pragma unroll
        for (int d = 0; d < 8; ++d) dst[d] = 0;
        return;
    }
    const int cbase[8] = {0, 4, 8, 12, 16, 20, 24, 28};
    float inv = 1.0f / qd_scale(amax[0]);
    const float* src = x + ((long)n * C_IN * H_IN + (yy - 1)) * W_IN + (xx - 1);
#pragma unroll
    for (int d = 0; d < 8; ++d) {
        int packed = 0;
#pragma unroll
        for (int b = 0; b < 4; ++b) {
            int c = cbase[d] + b;
            packed |= q8(src[(long)c * HW], inv) << (8 * b);
        }
        dst[d] = packed;
    }
}

__global__ void quantize_w_kernel(const float* __restrict__ w,
                                  const unsigned* __restrict__ amax,
                                  signed char* __restrict__ wq) {
    int t = blockIdx.x * blockDim.x + threadIdx.x;
    if (t >= 2560) return;
    int nt   = t / 1280;
    int rem  = t % 1280;
    int kc   = rem / 256;
    int rem2 = rem % 256;
    int lane = rem2 / 8;
    int v    = rem2 % 8;
    float inv = 1.0f / qd_scale(amax[1]);
    int o     = nt * 16 + (lane & 15);
    int kbase = kc * 64 + (v >> 2) * 32 + ((lane >> 4) * 16) + (v & 3) * 4;
    int packed = 0;
#pragma unroll
    for (int b = 0; b < 4; ++b) {
        int K = kbase + b;
        int tap = K >> 5, c = K & 31;
        int qb = 0;
        if (tap < 9) qb = q8(w[((long)(o * C_IN + c)) * 9 + tap], inv);
        packed |= qb << (8 * b);
    }
    ((int*)wq)[t] = packed;
}

__global__ __launch_bounds__(320) void conv_wmma_kernel(
        const signed char* __restrict__ xq, const signed char* __restrict__ wq,
        const float* __restrict__ bias, const unsigned* __restrict__ amax,
        float* __restrict__ out) {
    __shared__ int xpose[10 * 512];

    const int wave = threadIdx.x >> 5;
    const int lane = threadIdx.x & 31;
    const int y  = blockIdx.x;
    const int n  = blockIdx.y;
    const int x0 = wave << 4;

    const float s = qd_scale(amax[0]) * qd_scale(amax[1]);

    const int m      = lane & 15;
    const int half16 = (lane >> 4) << 4;
    const signed char* xb = xq + (long)n * (HP * WP * 32);
    const v8i* bfrag = (const v8i*)wq;

    __builtin_prefetch(wq, 0, 0);

    v8i acc0 = {};
    v8i acc1 = {};

#pragma unroll
    for (int kc = 0; kc < 5; ++kc) {
        const int t0 = 2 * kc, t1 = 2 * kc + 1;
        const int ky0 = t0 / 3, kx0 = t0 % 3;
        int4 a0 = *(const int4*)(xb +
                   ((long)(y + ky0) * WP + (x0 + m + kx0)) * 32 + half16);
        int4 a1 = {0, 0, 0, 0};
        if (t1 < 9) {
            const int ky1 = t1 / 3, kx1 = t1 % 3;
            a1 = *(const int4*)(xb +
                   ((long)(y + ky1) * WP + (x0 + m + kx1)) * 32 + half16);
        }
        v8i A;
        A[0] = a0.x; A[1] = a0.y; A[2] = a0.z; A[3] = a0.w;
        A[4] = a1.x; A[5] = a1.y; A[6] = a1.z; A[7] = a1.w;

        v8i B0 = bfrag[(0 * 5 + kc) * 32 + lane];
        v8i B1 = bfrag[(1 * 5 + kc) * 32 + lane];

        acc0 = __builtin_amdgcn_wmma_i32_16x16x64_iu8(true, A, true, B0, acc0,
                                                      false, false);
        acc1 = __builtin_amdgcn_wmma_i32_16x16x64_iu8(true, A, true, B1, acc1,
                                                      false, false);
    }

    int* my = xpose + wave * 512;
    const int half8 = (lane >> 4) << 3;
    {
        int* d0 = my + (lane & 15) * 16 + half8;
        int* d1 = my + ((lane & 15) + 16) * 16 + half8;
        int4 lo0 = {acc0[0], acc0[1], acc0[2], acc0[3]};
        int4 hi0 = {acc0[4], acc0[5], acc0[6], acc0[7]};
        int4 lo1 = {acc1[0], acc1[1], acc1[2], acc1[3]};
        int4 hi1 = {acc1[4], acc1[5], acc1[6], acc1[7]};
        *(int4*)(d0)     = lo0;
        *(int4*)(d0 + 4) = hi0;
        *(int4*)(d1)     = lo1;
        *(int4*)(d1 + 4) = hi1;
    }
    __syncthreads();

    const int og = lane >> 2;
    const int xs = (lane & 3) << 2;
    const long outb = ((long)n * O_OUT) * HW + (long)y * W_IN + x0;
#pragma unroll
    for (int i = 0; i < 4; ++i) {
        const int o = i * 8 + og;
        int4 a = *(const int4*)(my + o * 16 + xs);
        const float b = bias[o];
        float4 f;
        f.x = (float)a.x * s + b;
        f.y = (float)a.y * s + b;
        f.z = (float)a.z * s + b;
        f.w = (float)a.w * s + b;
        *(float4*)(out + outb + (long)o * HW + xs) = f;
    }
}

extern "C" void kernel_launch(void* const* d_in, const int* in_sizes, int n_in,
                              void* d_out, int out_size, void* d_ws,
                              size_t ws_size, hipStream_t stream) {
    const float* x    = (const float*)d_in[0];
    const float* w    = (const float*)d_in[1];
    const float* bias = (const float*)d_in[2];
    float* out        = (float*)d_out;

    unsigned*    amax = (unsigned*)d_ws;
    signed char* wq   = (signed char*)d_ws + WQ_OFF;
    signed char* xqp  = (signed char*)d_ws + XQ_OFF;

    zero_amax_kernel<<<1, 32, 0, stream>>>(amax);
    absmax_kernel<<<1024, 256, 0, stream>>>(x, NX, amax + 0);
    absmax_kernel<<<(NW + 255) / 256, 256, 0, stream>>>(w, NW, amax + 1);
    quantize_x_kernel<<<(NPIX_PAD + 255) / 256, 256, 0, stream>>>(x, amax, xqp);
    quantize_w_kernel<<<10, 256, 0, stream>>>(w, amax, wq);
    conv_wmma_kernel<<<dim3(160, 32), 320, 0, stream>>>(xqp, wq, bias, amax,
                                                        out);
}
